// Transformer_18657337934272
// MI455X (gfx1250) — matrix operand order repaired and checked by reading, not compiled, not run on hardware
//
#include <hip/hip_runtime.h>
#include <math.h>

// ---------------- problem constants (match reference) ----------------
#define Bn    32
#define S0c   288
#define Dm    1024
#define Hn    16
#define HDc   64
#define DFFc  4096
#define DLLMc 4096
#define NBLKc 6
#define WHALFc 3

typedef __attribute__((ext_vector_type(16))) __bf16 v16bf;
typedef __attribute__((ext_vector_type(8)))  float  v8f;

// ---------------- small device helpers ----------------
__device__ __forceinline__ unsigned short f2bf(float f) {
  unsigned int u = __float_as_uint(f);
  u += 0x7fffu + ((u >> 16) & 1u);       // round-to-nearest-even
  return (unsigned short)(u >> 16);
}
__device__ __forceinline__ float geluf(float x) {
  return 0.5f * x * (1.0f + erff(x * 0.70710678118654752f));   // exact gelu
}
__device__ __forceinline__ float wave_sum(float v) {
  #pragma unroll
  for (int o = 16; o > 0; o >>= 1) v += __shfl_xor(v, o, 32);
  return v;
}

// ---------------- weight convert + transpose: W[K][N] f32 -> Wt[N][K] bf16 --------
__global__ void k_wt(const float* __restrict__ W, unsigned short* __restrict__ Wt,
                     int K, int N, int total) {
  int idx = blockIdx.x * blockDim.x + threadIdx.x;
  if (idx >= total) return;
  int n = idx % N, k = idx / N;                 // coalesced read of W
  Wt[(size_t)n * K + k] = f2bf(W[idx]);
}

// ---------------- x = x_in + sinusoidal PE; also bf16 copy ----------------
__global__ void k_pe(const float* __restrict__ xin, float* __restrict__ x,
                     unsigned short* __restrict__ xbf, int total) {
  int idx = blockIdx.x * blockDim.x + threadIdx.x;
  if (idx >= total) return;
  int d = idx % Dm;
  int spos = (idx / Dm) % S0c;
  float ang = (float)spos * __expf(-(float)(d & ~1) * (9.210340371976184f / (float)Dm));
  float pe = (d & 1) ? __cosf(ang) : __sinf(ang);
  float v = xin[idx] + pe;
  x[idx] = v;
  xbf[idx] = f2bf(v);
}

// ---------------- bool mask -> float mask ----------------
__global__ void k_mask_init(const unsigned char* __restrict__ m, float* __restrict__ mf, int n) {
  int i = blockIdx.x * blockDim.x + threadIdx.x;
  if (i < n) mf[i] = m[i] ? 1.0f : 0.0f;
}
__global__ void k_copy(const float* __restrict__ a, float* __restrict__ b, int n) {
  int i = blockIdx.x * blockDim.x + threadIdx.x;
  if (i < n) b[i] = a[i];
}

// ---------------- WMMA GEMM: C[M][N] = A[M][K](bf16) @ Wt[N][K](bf16)^T + bias ----
// ep: 0 = none, 1 = exact gelu. scale (per column) applied after bias (for ls).
// Block: 256 threads = 8 waves; block tile 256(M) x 64(N); wave tile 32x64.
// B tile staged to double-buffered LDS via CDNA5 global_load_async_to_lds_b128
// (ASYNCcnt), one barrier per 32-K step.
// Requires M%256==0, N%64==0, K%32==0 (true for all shapes here).
__global__ __launch_bounds__(256) void
k_gemm(const unsigned short* __restrict__ A, const unsigned short* __restrict__ Wt,
       const float* __restrict__ bias, const float* __restrict__ scale,
       float* __restrict__ Cf, unsigned short* __restrict__ Cbf,
       int M, int N, int K, int ep) {
  __shared__ __align__(16) unsigned short lB[2][64 * 32];   // 2 x (64 N-rows x 32 K)

  const int tid  = threadIdx.x;
  const int wave = tid >> 5;
  const int lane = tid & 31;
  const bool hi  = lane >= 16;
  const int m0   = blockIdx.x * 256 + wave * 32;
  const int n0   = blockIdx.y * 64;

  v8f acc[8];
  #pragma unroll
  for (int t = 0; t < 8; ++t) acc[t] = (v8f){0.f,0.f,0.f,0.f,0.f,0.f,0.f,0.f};

  const unsigned short* Arow0 = A + (size_t)(m0 + (lane & 15)) * K;
  const unsigned short* Arow1 = Arow0 + (size_t)16 * K;

  const int sr = tid >> 2;          // 0..63 : N row staged by this thread
  const int sc_ = tid & 3;          // 0..3  : 8-element chunk
  const unsigned short* gB = Wt + (size_t)(n0 + sr) * K + sc_ * 8;

  // async direct-to-LDS staging of one 64x32 bf16 B tile (16B per thread)
  auto stage = [&](int k0, int buf) {
    unsigned lds = (unsigned)(size_t)&lB[buf][sr * 32 + sc_ * 8];
    unsigned long long ga = (unsigned long long)(size_t)(gB + k0);
    asm volatile("global_load_async_to_lds_b128 %0, %1, off"
                 :: "v"(lds), "v"(ga) : "memory");
  };

  stage(0, 0);
  asm volatile("s_wait_asynccnt 0" ::: "memory");
  __syncthreads();

  int buf = 0;
  for (int k0 = 0; k0 < K; k0 += 32) {
    const bool more = (k0 + 32) < K;          // uniform branch
    if (more) stage(k0 + 32, buf ^ 1);        // prefetch next tile into other buffer

    // A fragments per ISA layout: lanes<16: K {k0..k0+7, k0+16..k0+23}; lanes>=16: +8
    union { uint4 u[2]; v16bf v; } a0, a1;
    const int ka = k0 + (hi ? 8 : 0);
    a0.u[0] = *(const uint4*)(Arow0 + ka);
    a0.u[1] = *(const uint4*)(Arow0 + ka + 16);
    a1.u[0] = *(const uint4*)(Arow1 + ka);
    a1.u[1] = *(const uint4*)(Arow1 + ka + 16);
    __builtin_prefetch(Arow0 + ka + 32, 0, 3);
    __builtin_prefetch(Arow1 + ka + 32, 0, 3);

    const int kb = hi ? 8 : 0;    // B: lanes<16 hold K {k0..k0+7, k0+16..k0+23}; lanes>=16: +8
    #pragma unroll
    for (int t = 0; t < 4; ++t) {
      union { uint4 u[2]; v16bf v; } bfr;
      const unsigned short* brow = &lB[buf][(t * 16 + (lane & 15)) * 32 + kb];
      bfr.u[0] = *(const uint4*)(brow);
      bfr.u[1] = *(const uint4*)(brow + 16);
      acc[t] = __builtin_amdgcn_wmma_f32_16x16x32_bf16(
          false, a0.v, false, bfr.v, (short)0, acc[t], false, false);
      acc[4 + t] = __builtin_amdgcn_wmma_f32_16x16x32_bf16(
          false, a1.v, false, bfr.v, (short)0, acc[4 + t], false, false);
    }

    if (more) asm volatile("s_wait_asynccnt 0" ::: "memory");  // own async done
    __syncthreads();                                           // everyone's async done
    buf ^= 1;
  }

  // epilogue: VGPR e -> row base+e (lanes 0-15) / base+8+e (lanes 16-31), col = lane%16
  const int colLane = lane & 15;
  #pragma unroll
  for (int g2i = 0; g2i < 2; ++g2i) {
    const int rowBase = m0 + g2i * 16 + (hi ? 8 : 0);
    #pragma unroll
    for (int t = 0; t < 4; ++t) {
      const int col = n0 + t * 16 + colLane;
      const float bc = bias ? bias[col] : 0.0f;
      const float sc2 = scale ? scale[col] : 1.0f;
      #pragma unroll
      for (int e = 0; e < 8; ++e) {
        float v = acc[g2i * 4 + t][e] + bc;
        if (ep == 1) v = geluf(v);
        v *= sc2;
        const size_t idx = (size_t)(rowBase + e) * N + col;
        if (Cf)  Cf[idx]  = v;
        if (Cbf) Cbf[idx] = f2bf(v);
      }
    }
  }
}

// ---------------- attention: wave per (b,h,q); RoPE fused; 7-tap local softmax ----
__global__ __launch_bounds__(256) void
k_attn(const float* __restrict__ q, const float* __restrict__ k,
       const float* __restrict__ v, const float* __restrict__ maskf,
       unsigned short* __restrict__ obf, int s) {
  const int wid  = blockIdx.x * (blockDim.x >> 5) + (threadIdx.x >> 5);
  const int lane = threadIdx.x & 31;
  const int total = Bn * Hn * s;
  if (wid >= total) return;
  const int b = wid / (Hn * s);
  const int r = wid % (Hn * s);
  const int h = r / s;
  const int qpos = r % s;

  // inv freq for rope pair i = lane (output dims lane and lane+32)
  const float inv = __powf(10000.0f, -2.0f * (float)lane / (float)HDc);
  const float aq = (float)qpos * inv;
  const float cq = __cosf(aq), sq = __sinf(aq);
  const float* qp = q + ((size_t)(b * s + qpos)) * Dm + h * HDc;
  const float q0 = qp[2 * lane], q1 = qp[2 * lane + 1];
  const float qa = q0 * cq - q1 * sq;     // rotated dim lane
  const float qb = q1 * cq + q0 * sq;     // rotated dim lane+32

  float scv[7];
  #pragma unroll
  for (int t = 0; t < 7; ++t) {
    const int j = qpos + t - WHALFc;
    float sval = -INFINITY;
    if (j >= 0 && j < s) {
      const bool ok = (maskf[b * s + j] > 0.0f) || (j == qpos);  // local&mask | eye
      if (ok) {
        const float* kp = k + ((size_t)(b * s + j)) * Dm + h * HDc;
        const float ak = (float)j * inv;
        const float ck = __cosf(ak), sk = __sinf(ak);
        const float k0 = kp[2 * lane], k1 = kp[2 * lane + 1];
        const float kra = k0 * ck - k1 * sk;
        const float krb = k1 * ck + k0 * sk;
        float p = qa * kra + qb * krb;
        p = wave_sum(p);                 // uniform branch -> safe
        sval = p * 0.125f;               // 1/sqrt(64)
      }
    }
    scv[t] = sval;
  }

  float m = scv[0];
  #pragma unroll
  for (int t = 1; t < 7; ++t) m = fmaxf(m, scv[t]);
  float ev[7], sum = 0.0f;
  #pragma unroll
  for (int t = 0; t < 7; ++t) { ev[t] = __expf(scv[t] - m); sum += ev[t]; }
  const float rinv = 1.0f / sum;

  float o0 = 0.0f, o1 = 0.0f;
  #pragma unroll
  for (int t = 0; t < 7; ++t) {
    const int j = qpos + t - WHALFc;
    if (j >= 0 && j < s) {
      const float w = ev[t] * rinv;
      const float* vp = v + ((size_t)(b * s + j)) * Dm + h * HDc;
      o0 += w * vp[lane];
      o1 += w * vp[lane + 32];
    }
  }
  unsigned short* op = obf + ((size_t)(b * s + qpos)) * Dm + h * HDc;
  op[lane]      = f2bf(o0);
  op[lane + 32] = f2bf(o1);
}

// ---------------- x = LN(x + delta)*g + be, also bf16 copy; wave per token -------
__global__ __launch_bounds__(256) void
k_res_ln(float* __restrict__ x, const float* __restrict__ dlt,
         const float* __restrict__ g, const float* __restrict__ be,
         unsigned short* __restrict__ xbf, int M) {
  const int wid  = blockIdx.x * 8 + (threadIdx.x >> 5);
  const int lane = threadIdx.x & 31;
  if (wid >= M) return;
  const size_t off = (size_t)wid * Dm;
  float rr[Dm / 32];
  float sum = 0.0f;
  #pragma unroll
  for (int i = 0; i < Dm / 32; ++i) {
    const float t = x[off + i * 32 + lane] + dlt[off + i * 32 + lane];
    rr[i] = t; sum += t;
  }
  sum = wave_sum(sum);
  const float mu = sum / (float)Dm;
  float vs = 0.0f;
  #pragma unroll
  for (int i = 0; i < Dm / 32; ++i) { const float d = rr[i] - mu; vs += d * d; }
  vs = wave_sum(vs);
  const float rs = rsqrtf(vs / (float)Dm + 1e-5f);
  #pragma unroll
  for (int i = 0; i < Dm / 32; ++i) {
    const int c = i * 32 + lane;
    const float y = (rr[i] - mu) * rs * g[c] + be[c];
    x[off + c] = y;
    xbf[off + c] = f2bf(y);
  }
}

// ---------------- downsample: window 3, stride 2, pad (1,1); + mask --------------
__global__ void k_ds(const float* __restrict__ xin, const float* __restrict__ min_,
                     float* __restrict__ xout, float* __restrict__ mout,
                     unsigned short* __restrict__ xbf, int sin_, int sout, int total) {
  int idx = blockIdx.x * blockDim.x + threadIdx.x;
  if (idx >= total) return;
  const int d  = idx % Dm;
  const int so = (idx / Dm) % sout;
  const int b  = idx / (Dm * sout);
  float sum = 0.0f;
  #pragma unroll
  for (int t = -1; t <= 1; ++t) {
    const int j = 2 * so + t;
    if (j >= 0 && j < sin_) sum += xin[((size_t)(b * sin_ + j)) * Dm + d];
  }
  const float val = sum * (1.0f / 3.0f);
  const size_t oi = ((size_t)(b * sout + so)) * Dm + d;
  xout[oi] = val;
  xbf[oi]  = f2bf(val);
  if (d == 0) {
    float ms = 0.0f;
    #pragma unroll
    for (int t = -1; t <= 1; ++t) {
      const int j = 2 * so + t;
      if (j >= 0 && j < sin_) ms += min_[b * sin_ + j];
    }
    mout[b * sout + so] = (ms > 0.0f || so == 0) ? 1.0f : 0.0f;
  }
}

// ================================ host orchestration =============================
extern "C" void kernel_launch(void* const* d_in, const int* in_sizes, int n_in,
                              void* d_out, int out_size, void* d_ws, size_t ws_size,
                              hipStream_t stream) {
  const float*         x_in   = (const float*)d_in[0];
  const unsigned char* mask_in= (const unsigned char*)d_in[1];
  const float* Wq  = (const float*)d_in[2];
  const float* bq  = (const float*)d_in[3];
  const float* Wk  = (const float*)d_in[4];
  const float* bk  = (const float*)d_in[5];
  const float* Wv  = (const float*)d_in[6];
  const float* bv  = (const float*)d_in[7];
  const float* Wo  = (const float*)d_in[8];
  const float* bo  = (const float*)d_in[9];
  const float* ls  = (const float*)d_in[10];
  const float* W1  = (const float*)d_in[11];
  const float* b1  = (const float*)d_in[12];
  const float* W2  = (const float*)d_in[13];
  const float* b2  = (const float*)d_in[14];
  const float* g1  = (const float*)d_in[15];
  const float* be1 = (const float*)d_in[16];
  const float* g2  = (const float*)d_in[17];
  const float* be2 = (const float*)d_in[18];
  const float* Wout= (const float*)d_in[19];
  const float* bout= (const float*)d_in[20];

  // ------- workspace layout -------
  char* base = (char*)d_ws;
  size_t off = 0;
  auto alloc = [&](size_t bytes) -> char* {
    char* p = base + off;
    off = (off + bytes + 255) & ~(size_t)255;
    return p;
  };
  const size_t DD = (size_t)Dm * Dm;       // 1M
  const size_t DF = (size_t)Dm * DFFc;     // 4M
  const size_t WPB = 4 * DD + 2 * DF;      // bf16 weights per block
  const size_t M0 = (size_t)Bn * S0c;      // 9216 tokens max

  unsigned short* wbf  = (unsigned short*)alloc((NBLKc * WPB + (size_t)Dm * DLLMc) * 2);
  float*          x0   = (float*)alloc(M0 * Dm * 4);
  float*          x1   = (float*)alloc(M0 * Dm * 4);
  unsigned short* xbf  = (unsigned short*)alloc(M0 * Dm * 2);
  float*          qf   = (float*)alloc(M0 * Dm * 4);
  float*          kf   = (float*)alloc(M0 * Dm * 4);
  float*          vf   = (float*)alloc(M0 * Dm * 4);
  unsigned short* obf  = (unsigned short*)alloc(M0 * Dm * 2);
  unsigned short* h1bf = (unsigned short*)alloc(M0 * DFFc * 2);
  float*          mskA = (float*)alloc((size_t)Bn * S0c * 4);
  float*          mskB = (float*)alloc((size_t)Bn * S0c * 4);

  // ------- weight conversion (fp32 -> transposed bf16) -------
  auto conv = [&](const float* W, unsigned short* Wt, int K, int N) {
    int total = K * N;
    k_wt<<<(total + 255) / 256, 256, 0, stream>>>(W, Wt, K, N, total);
  };
  for (int i = 0; i < NBLKc; ++i) {
    unsigned short* wb = wbf + (size_t)i * WPB;
    conv(Wq + (size_t)i * DD, wb + 0 * DD, Dm, Dm);
    conv(Wk + (size_t)i * DD, wb + 1 * DD, Dm, Dm);
    conv(Wv + (size_t)i * DD, wb + 2 * DD, Dm, Dm);
    conv(Wo + (size_t)i * DD, wb + 3 * DD, Dm, Dm);
    conv(W1 + (size_t)i * DF, wb + 4 * DD, Dm, DFFc);       // -> [DFF][D]
    conv(W2 + (size_t)i * DF, wb + 4 * DD + DF, DFFc, Dm);  // -> [D][DFF]
  }
  unsigned short* woutT = wbf + (size_t)NBLKc * WPB;
  conv(Wout, woutT, Dm, DLLMc);                              // -> [DLLM][D]

  // ------- positional encoding + mask init -------
  {
    int tot = Bn * S0c * Dm;
    k_pe<<<(tot + 255) / 256, 256, 0, stream>>>(x_in, x0, xbf, tot);
    k_mask_init<<<(Bn * S0c + 255) / 256, 256, 0, stream>>>(mask_in, mskA, Bn * S0c);
  }

  auto gemm = [&](const unsigned short* A, const unsigned short* Wt, const float* bias,
                  const float* scale, float* Cf, unsigned short* Cbf,
                  int M, int N, int K, int ep) {
    dim3 grid(M / 256, N / 64);
    k_gemm<<<grid, 256, 0, stream>>>(A, Wt, bias, scale, Cf, Cbf, M, N, K, ep);
  };

  // ------- transformer blocks -------
  float* xcur = x0; float* xalt = x1;
  float* mcur = mskA; float* malt = mskB;
  int s = S0c;
  for (int i = 0; i < NBLKc; ++i) {
    const int M = Bn * s;
    unsigned short* wb = wbf + (size_t)i * WPB;
    // q,k,v projections (bf16 WMMA, f32 out)
    gemm(xbf, wb + 0 * DD, bq + (size_t)i * Dm, nullptr, qf, nullptr, M, Dm, Dm, 0);
    gemm(xbf, wb + 1 * DD, bk + (size_t)i * Dm, nullptr, kf, nullptr, M, Dm, Dm, 0);
    gemm(xbf, wb + 2 * DD, bv + (size_t)i * Dm, nullptr, vf, nullptr, M, Dm, Dm, 0);
    // local-window attention with fused RoPE -> bf16 o
    {
      const int waves = Bn * Hn * s;
      k_attn<<<(waves + 7) / 8, 256, 0, stream>>>(qf, kf, vf, mcur, obf, s);
    }
    // a = (o @ Wo + bo) * ls   (reuse qf as f32 scratch)
    gemm(obf, wb + 3 * DD, bo + (size_t)i * Dm, ls + (size_t)i * Dm, qf, nullptr, M, Dm, Dm, 0);
    // x = LN(x + a)
    k_res_ln<<<M / 8, 256, 0, stream>>>(xcur, qf, g1 + (size_t)i * Dm, be1 + (size_t)i * Dm, xbf, M);
    // h1 = gelu(x @ W1 + b1) -> bf16 only
    gemm(xbf, wb + 4 * DD, b1 + (size_t)i * DFFc, nullptr, nullptr, h1bf, M, DFFc, Dm, 1);
    // f = h1 @ W2 + b2   (reuse kf as f32 scratch)
    gemm(h1bf, wb + 4 * DD + DF, b2 + (size_t)i * Dm, nullptr, kf, nullptr, M, Dm, DFFc, 0);
    // x = LN(x + f)
    k_res_ln<<<M / 8, 256, 0, stream>>>(xcur, kf, g2 + (size_t)i * Dm, be2 + (size_t)i * Dm, xbf, M);
    // downsample after blocks 1 and 3
    if (i == 1 || i == 3) {
      const int so = s / 2;
      const int tot = Bn * so * Dm;
      k_ds<<<(tot + 255) / 256, 256, 0, stream>>>(xcur, mcur, xalt, malt, xbf, s, so, tot);
      float* t = xcur; xcur = xalt; xalt = t;
      float* tm = mcur; mcur = malt; malt = tm;
      s = so;
    }
  }

  // ------- output head + mask -------
  const int Mf = Bn * s;  // 32*72 = 2304
  gemm(xbf, woutT, bout, nullptr, (float*)d_out, nullptr, Mf, DLLMc, Dm, 0);
  k_copy<<<(Mf + 255) / 256, 256, 0, stream>>>(mcur, (float*)d_out + (size_t)Mf * DLLMc, Mf);
}
